// Transformer_2319282340622
// MI455X (gfx1250) — matrix operand order repaired and checked by reading, not compiled, not run on hardware
//
#include <hip/hip_runtime.h>
#include <math.h>

typedef __attribute__((ext_vector_type(16))) _Float16 v16h;
typedef __attribute__((ext_vector_type(8)))  _Float16 v8h;
typedef __attribute__((ext_vector_type(8)))  float    v8f;
typedef __attribute__((ext_vector_type(4)))  float    v4f;
typedef unsigned int u32x4 __attribute__((ext_vector_type(4)));
typedef int          i32x4 __attribute__((ext_vector_type(4)));
typedef int          i32x8 __attribute__((ext_vector_type(8)));

#define N_LAYERS 4
#define BATCH 2
#define SEQ   1024
#define DMODEL 1024
#define NHEAD 16
#define HEADD 64
#define FFDIM 4096
#define VOCAB 50257

#if defined(__has_builtin)
#if __has_builtin(__builtin_amdgcn_tensor_load_to_lds) && __has_builtin(__builtin_amdgcn_s_wait_tensorcnt)
#define USE_TDM 1
#endif
#endif
#ifndef USE_TDM
#define USE_TDM 0
#endif

__device__ __forceinline__ v8f wmma16(v16h a, v16h b, v8f c) {
  return __builtin_amdgcn_wmma_f32_16x16x32_f16(false, a, false, b, (short)0, c, false, false);
}

__device__ __forceinline__ float gelu_tanh(float x) {
  float x3 = x * x * x;
  return 0.5f * x * (1.0f + tanhf(0.7978845608f * (x + 0.044715f * x3)));
}

#if USE_TDM
// Issue one 2D TDM tile load: 128 rows x 32 f16 (K-contiguous) -> LDS.
// D# per CDNA5 ISA.
__device__ __forceinline__ void tdm_load_2d(unsigned lds_addr, const _Float16* gptr,
                                            unsigned tdim0, unsigned tdim1, unsigned stride0) {
  unsigned long long ga = (unsigned long long)(uintptr_t)gptr;
  u32x4 g0;
  g0[0] = 1u;                                   // count=1, user descriptor
  g0[1] = lds_addr;                             // lds_addr (bytes)
  g0[2] = (unsigned)ga;                         // global_addr[31:0]
  g0[3] = (unsigned)((ga >> 32) & 0x01FFFFFFu)  // global_addr[56:32]
        | (2u << 30);                           // type=2 ("image")
  i32x8 g1;
  g1[0] = (int)(1u << 16);                      // workgroup_mask=0, data_size=1 (2 bytes)
  g1[1] = (int)((tdim0 & 0xFFFFu) << 16);                                   // tensor_dim0[15:0]
  g1[2] = (int)(((tdim0 >> 16) & 0xFFFFu) | ((tdim1 & 0xFFFFu) << 16));     // td0[31:16] | td1[15:0]
  g1[3] = (int)(((tdim1 >> 16) & 0xFFFFu) | (32u << 16));                   // td1[31:16] | tile_dim0=32
  g1[4] = (int)128u;                            // tile_dim1=128, tile_dim2=0
  g1[5] = (int)stride0;                         // tensor_dim0_stride[31:0] (elements)
  g1[6] = 0;                                    // stride0[47:32]=0, stride1[15:0]=0
  g1[7] = 0;
  i32x4 z4 = {0, 0, 0, 0};
#if __clang_major__ >= 23
  i32x8 z8 = {0, 0, 0, 0, 0, 0, 0, 0};
  __builtin_amdgcn_tensor_load_to_lds(g0, g1, z4, z4, z8, 0);
#else
  __builtin_amdgcn_tensor_load_to_lds(g0, g1, z4, z4, 0);
#endif
}
#endif

// ---------------------------------------------------------------- embedding
__global__ __launch_bounds__(256)
void embed_kernel(const int* __restrict__ tokens, const float* __restrict__ wte,
                  float* __restrict__ x, int D) {
  int row = blockIdx.x;
  int tok = tokens[row];
  const float* src = wte + (size_t)tok * D;
  float* dst = x + (size_t)row * D;
  for (int i = threadIdx.x; i < D; i += 256) dst[i] = src[i];
}

// ---------------------------------------------------------------- layernorm (f32 in -> f16 out)
__global__ __launch_bounds__(256)
void ln_kernel(const float* __restrict__ x, const float* __restrict__ gamma,
               const float* __restrict__ beta, _Float16* __restrict__ y, int D) {
  __shared__ float rs[8], rq[8], stat[2];
  int row = blockIdx.x;
  const float* xr = x + (size_t)row * D;
  float sum = 0.f, sq = 0.f;
  for (int i = threadIdx.x; i < D; i += 256) {
    float v = xr[i];
    sum += v; sq += v * v;
  }
  for (int o = 16; o > 0; o >>= 1) {
    sum += __shfl_down(sum, o, 32);
    sq  += __shfl_down(sq,  o, 32);
  }
  int wave = threadIdx.x >> 5, lane = threadIdx.x & 31;
  if (lane == 0) { rs[wave] = sum; rq[wave] = sq; }
  __syncthreads();
  if (threadIdx.x == 0) {
    float s = 0.f, q2 = 0.f;
    for (int i = 0; i < 8; ++i) { s += rs[i]; q2 += rq[i]; }
    float m = s / (float)D;
    stat[0] = m;
    stat[1] = q2 / (float)D - m * m;
  }
  __syncthreads();
  float mean = stat[0];
  float inv = rsqrtf(stat[1] + 1e-6f);
  _Float16* yr = y + (size_t)row * D;
  for (int i = threadIdx.x; i < D; i += 256)
    yr[i] = (_Float16)((xr[i] - mean) * inv * gamma[i] + beta[i]);
}

// ----------------------------------------------- weight convert: f32 [K][N] -> f16 [N][K]
__global__ __launch_bounds__(256)
void wconv_t(const float* __restrict__ W, _Float16* __restrict__ Wt, int K, int N) {
  __shared__ float t[32][33];
  int k0 = blockIdx.x * 32, n0 = blockIdx.y * 32;
  int tx = threadIdx.x & 31, ty = threadIdx.x >> 5;  // 32 x 8
  #pragma unroll
  for (int i = 0; i < 32; i += 8)
    t[ty + i][tx] = W[(size_t)(k0 + ty + i) * N + (n0 + tx)];
  __syncthreads();
  #pragma unroll
  for (int i = 0; i < 32; i += 8)
    Wt[(size_t)(n0 + ty + i) * K + (k0 + tx)] = (_Float16)t[tx][ty + i];
}

// ----------------------------------------------- plain convert f32 -> f16 (wte is already [N][K])
__global__ __launch_bounds__(256)
void conv16(const float* __restrict__ in, _Float16* __restrict__ out, size_t n) {
  size_t i = (size_t)blockIdx.x * 256 + threadIdx.x;
  size_t st = (size_t)gridDim.x * 256;
  for (; i < n; i += st) out[i] = (_Float16)in[i];
}

// ---------------------------------------------------------------- GEMM (all-f16 operands)
// A: f16 [M][K] row-major.  B: f16 [N][K] (pre-transposed).  acc f32.
// EPI: 0 = bias -> f16 out; 1 = bias+gelu -> f16 out; 2 = bias+residual -> f32 out; 3 = plain f32 out.
template<int EPI>
__global__ __launch_bounds__(256)
void gemm_h16(const _Float16* __restrict__ A, const _Float16* __restrict__ B,
              const float* __restrict__ bias, void* __restrict__ Cv,
              const float* __restrict__ resid, int M, int N, int K, int ldc) {
#if USE_TDM
  __shared__ __align__(16) _Float16 As[2][128][32];
  __shared__ __align__(16) _Float16 Bs[2][128][32];
#else
  __shared__ __align__(16) _Float16 As[1][128][32];
  __shared__ __align__(16) _Float16 Bs[1][128][32];
#endif
  const int tid  = threadIdx.x;
  const int lane = tid & 31;
  const int wave = tid >> 5;
  const int wm   = wave & 1;
  const int wn   = wave >> 1;
  const int half = lane >> 4;
  const int l16  = lane & 15;
  const int m0   = blockIdx.x * 128;
  const int n0   = blockIdx.y * 128;
  const int nk   = K >> 5;

  const _Float16* Abase = A + (size_t)m0 * K;
  const _Float16* Bbase = B + (size_t)n0 * K;

  v8f acc[4][2];
  #pragma unroll
  for (int i = 0; i < 4; ++i)
    #pragma unroll
    for (int j = 0; j < 2; ++j)
      #pragma unroll
      for (int e = 0; e < 8; ++e) acc[i][j][e] = 0.f;

#if USE_TDM
  const unsigned ldsA[2] = { (unsigned)(uintptr_t)&As[0][0][0], (unsigned)(uintptr_t)&As[1][0][0] };
  const unsigned ldsB[2] = { (unsigned)(uintptr_t)&Bs[0][0][0], (unsigned)(uintptr_t)&Bs[1][0][0] };
  const unsigned rowsA = (unsigned)(M - m0);
  const unsigned rowsB = (unsigned)(N - n0);   // descriptor clips ragged N edge (OOB reads -> 0)
  if (wave == 0) {
    tdm_load_2d(ldsA[0], Abase, (unsigned)K, rowsA, (unsigned)K);
    tdm_load_2d(ldsB[0], Bbase, (unsigned)K, rowsB, (unsigned)K);
  }
#endif

  for (int t = 0; t < nk; ++t) {
#if USE_TDM
    const int cur = t & 1;
    if (wave == 0) {
      if (t + 1 < nk) {
        const int k1 = (t + 1) << 5;
        tdm_load_2d(ldsA[cur ^ 1], Abase + k1, (unsigned)(K - k1), rowsA, (unsigned)K);
        tdm_load_2d(ldsB[cur ^ 1], Bbase + k1, (unsigned)(K - k1), rowsB, (unsigned)K);
        __builtin_amdgcn_s_wait_tensorcnt(2);  // current tile's 2 ops done; next 2 in flight
      } else {
        __builtin_amdgcn_s_wait_tensorcnt(0);
      }
    }
    __syncthreads();
#else
    const int cur = 0;
    const int k0 = t << 5;
    {
      int r  = tid >> 2;             // 0..63
      int c8 = (tid & 3) * 8;        // 0,8,16,24
      #pragma unroll
      for (int p = 0; p < 2; ++p) {
        int row = r + p * 64;
        *(v8h*)&As[0][row][c8] = *(const v8h*)(Abase + (size_t)row * K + k0 + c8);
        v8h bz;
        #pragma unroll
        for (int e = 0; e < 8; ++e) bz[e] = (_Float16)0.f;
        if (n0 + row < N) bz = *(const v8h*)(Bbase + (size_t)row * K + k0 + c8);
        *(v8h*)&Bs[0][row][c8] = bz;
      }
    }
    __syncthreads();
#endif

    // --- fragments + WMMA
    v16h af[4];
    #pragma unroll
    for (int mt = 0; mt < 4; ++mt) {
      int r = wm * 64 + mt * 16 + l16;
      v8h lo = *(const v8h*)&As[cur][r][half * 8];
      v8h hi = *(const v8h*)&As[cur][r][16 + half * 8];
      #pragma unroll
      for (int e = 0; e < 8; ++e) { af[mt][e] = lo[e]; af[mt][8 + e] = hi[e]; }
    }
    v16h bf[2];
    #pragma unroll
    for (int nt = 0; nt < 2; ++nt) {
      int n = wn * 32 + nt * 16 + l16;
      v8h b0 = *(const v8h*)&Bs[cur][n][half * 8];
      v8h b1 = *(const v8h*)&Bs[cur][n][16 + half * 8];
      #pragma unroll
      for (int e = 0; e < 8; ++e) { bf[nt][e] = b0[e]; bf[nt][8 + e] = b1[e]; }
    }
    #pragma unroll
    for (int mt = 0; mt < 4; ++mt)
      #pragma unroll
      for (int nt = 0; nt < 2; ++nt)
        acc[mt][nt] = wmma16(af[mt], bf[nt], acc[mt][nt]);

    __syncthreads();
  }

  // --- epilogue
  float*    C32 = (float*)Cv;
  _Float16* C16 = (_Float16*)Cv;
  #pragma unroll
  for (int mt = 0; mt < 4; ++mt) {
    #pragma unroll
    for (int nt = 0; nt < 2; ++nt) {
      int gn = n0 + wn * 32 + nt * 16 + l16;
      if (gn >= N) continue;
      float bval = bias ? bias[gn] : 0.f;
      #pragma unroll
      for (int vv = 0; vv < 8; ++vv) {
        int gm = m0 + wm * 64 + mt * 16 + half * 8 + vv;
        if (gm >= M) continue;
        float r = acc[mt][nt][vv] + bval;
        if (EPI == 1) r = gelu_tanh(r);
        if (EPI == 2) r += resid[(size_t)gm * ldc + gn];
        if (EPI == 0 || EPI == 1) C16[(size_t)gm * ldc + gn] = (_Float16)r;
        else                      C32[(size_t)gm * ldc + gn] = r;
      }
    }
  }
}

// ------------------------------- logits fallback: A f16 [M][K], B f32 [N][K] (wte), f32 out
__global__ __launch_bounds__(256)
void gemm_logits_f32w(const _Float16* __restrict__ A, const float* __restrict__ B,
                      float* __restrict__ C, int M, int N, int K, int ldc) {
  __shared__ __align__(16) _Float16 Ls[128][32];
  __shared__ __align__(16) _Float16 Rs[128][32];
  const int tid = threadIdx.x, lane = tid & 31, wave = tid >> 5;
  const int wm = wave & 1, wn = wave >> 1, half = lane >> 4, l16 = lane & 15;
  const int m0 = blockIdx.x * 128, n0 = blockIdx.y * 128;

  v8f acc[4][2];
  #pragma unroll
  for (int i = 0; i < 4; ++i)
    #pragma unroll
    for (int j = 0; j < 2; ++j)
      #pragma unroll
      for (int e = 0; e < 8; ++e) acc[i][j][e] = 0.f;

  for (int k0 = 0; k0 < K; k0 += 32) {
    {
      int r = tid >> 2, c8 = (tid & 3) * 8;
      #pragma unroll
      for (int p = 0; p < 2; ++p) {
        int row = r + p * 64;
        *(v8h*)&Ls[row][c8] = *(const v8h*)(A + (size_t)(m0 + row) * K + k0 + c8);
      }
      int rr = tid >> 3, c4 = (tid & 7) * 4;
      #pragma unroll
      for (int p = 0; p < 4; ++p) {
        int row = rr + p * 32;
        int gn = n0 + row;
        v4f b4 = {0.f, 0.f, 0.f, 0.f};
        if (gn < N) b4 = *(const v4f*)(B + (size_t)gn * K + k0 + c4);
        #pragma unroll
        for (int i = 0; i < 4; ++i) Rs[row][c4 + i] = (_Float16)b4[i];
      }
    }
    __syncthreads();
    v16h af[4];
    #pragma unroll
    for (int mt = 0; mt < 4; ++mt) {
      int r = wm * 64 + mt * 16 + l16;
      v8h lo = *(const v8h*)&Ls[r][half * 8];
      v8h hi = *(const v8h*)&Ls[r][16 + half * 8];
      #pragma unroll
      for (int e = 0; e < 8; ++e) { af[mt][e] = lo[e]; af[mt][8 + e] = hi[e]; }
    }
    v16h bf[2];
    #pragma unroll
    for (int nt = 0; nt < 2; ++nt) {
      int n = wn * 32 + nt * 16 + l16;
      v8h b0 = *(const v8h*)&Rs[n][half * 8];
      v8h b1 = *(const v8h*)&Rs[n][16 + half * 8];
      #pragma unroll
      for (int e = 0; e < 8; ++e) { bf[nt][e] = b0[e]; bf[nt][8 + e] = b1[e]; }
    }
    #pragma unroll
    for (int mt = 0; mt < 4; ++mt)
      #pragma unroll
      for (int nt = 0; nt < 2; ++nt)
        acc[mt][nt] = wmma16(af[mt], bf[nt], acc[mt][nt]);
    __syncthreads();
  }
  #pragma unroll
  for (int mt = 0; mt < 4; ++mt)
    #pragma unroll
    for (int nt = 0; nt < 2; ++nt) {
      int gn = n0 + wn * 32 + nt * 16 + l16;
      if (gn >= N) continue;
      #pragma unroll
      for (int vv = 0; vv < 8; ++vv) {
        int gm = m0 + wm * 64 + mt * 16 + half * 8 + vv;
        if (gm < M) C[(size_t)gm * ldc + gn] = acc[mt][nt][vv];
      }
    }
}

// ---------------------------------------------------------------- attention (f16 q/k/v, f16 out)
#define ATTN_SMEM (16 * 1024 * 4 + 16 * 1024 * 2 + 64 * 32 * 2)  // 102400 B
__global__ __launch_bounds__(32)
void attn_kernel(const _Float16* __restrict__ q, const _Float16* __restrict__ k,
                 const _Float16* __restrict__ v, _Float16* __restrict__ o) {
  extern __shared__ char smraw[];
  float*    sc = (float*)smraw;                 // [16][1024] scores (f32)
  _Float16* pm = (_Float16*)(sc + 16 * 1024);   // [16][1024] probs (f16)
  _Float16* vt = pm + 16 * 1024;                // [64][32]   v tile transposed

  const int lane = threadIdx.x;
  const int half = lane >> 4;
  const int l16  = lane & 15;
  const int qt = blockIdx.x, hh = blockIdx.y, bb = blockIdx.z;
  const float scale = 0.125f;
  const float slope = exp2f(-0.5f * (float)(hh + 1));

  v16h aq[2];
  {
    int qi = qt * 16 + l16;
    const _Float16* qrow = q + (((size_t)bb * SEQ + qi) * NHEAD + hh) * HEADD;
    #pragma unroll
    for (int d0 = 0; d0 < 2; ++d0) {
      v8h lo = *(const v8h*)(qrow + d0 * 32 + half * 8);
      v8h hi = *(const v8h*)(qrow + d0 * 32 + 16 + half * 8);
      #pragma unroll
      for (int e = 0; e < 8; ++e) { aq[d0][e] = lo[e]; aq[d0][8 + e] = hi[e]; }
    }
  }

  for (int jt = 0; jt < SEQ / 16; ++jt) {
    int kj = jt * 16 + l16;
    const _Float16* krow = k + (((size_t)bb * SEQ + kj) * NHEAD + hh) * HEADD;
    v8f c;
    #pragma unroll
    for (int e = 0; e < 8; ++e) c[e] = 0.f;
    #pragma unroll
    for (int d0 = 0; d0 < 2; ++d0) {
      v16h bk;
      v8h b0 = *(const v8h*)(krow + d0 * 32 + half * 8);
      v8h b1 = *(const v8h*)(krow + d0 * 32 + 16 + half * 8);
      #pragma unroll
      for (int e = 0; e < 8; ++e) { bk[e] = b0[e]; bk[8 + e] = b1[e]; }
      c = wmma16(aq[d0], bk, c);
    }
    #pragma unroll
    for (int vv = 0; vv < 8; ++vv) {
      int m  = half * 8 + vv;
      int qi = qt * 16 + m;
      int j  = jt * 16 + l16;
      float s = c[vv] * scale - slope * (float)(SEQ - 1 - j);
      if (j > qi) s = -3.0e38f;
      sc[m * 1024 + j] = s;
    }
  }

  {
    int row = l16, cb = half * 512;
    float mx = -3.4e38f;
    for (int i = 0; i < 512; ++i) mx = fmaxf(mx, sc[row * 1024 + cb + i]);
    mx = fmaxf(mx, __shfl_xor(mx, 16, 32));
    float sum = 0.f;
    for (int i = 0; i < 512; ++i) sum += __expf(sc[row * 1024 + cb + i] - mx);
    sum += __shfl_xor(sum, 16, 32);
    float inv = 1.0f / sum;
    for (int i = 0; i < 512; ++i)
      pm[row * 1024 + cb + i] = (_Float16)(__expf(sc[row * 1024 + cb + i] - mx) * inv);
  }

  v8f oc[4];
  #pragma unroll
  for (int nt = 0; nt < 4; ++nt)
    #pragma unroll
    for (int e = 0; e < 8; ++e) oc[nt][e] = 0.f;

  for (int j0 = 0; j0 < SEQ; j0 += 32) {
    const _Float16* vrow = v + (((size_t)bb * SEQ + j0 + lane) * NHEAD + hh) * HEADD;
    #pragma unroll
    for (int c8 = 0; c8 < 8; ++c8) {
      v8h tv = *(const v8h*)(vrow + c8 * 8);
      #pragma unroll
      for (int e = 0; e < 8; ++e) vt[(c8 * 8 + e) * 32 + lane] = tv[e];
    }
    v16h ap;
    {
      v8h p0 = *(const v8h*)&pm[l16 * 1024 + j0 + half * 8];
      v8h p1 = *(const v8h*)&pm[l16 * 1024 + j0 + 16 + half * 8];
      #pragma unroll
      for (int e = 0; e < 8; ++e) { ap[e] = p0[e]; ap[8 + e] = p1[e]; }
    }
    #pragma unroll
    for (int nt = 0; nt < 4; ++nt) {
      v16h bv;
      v8h b0 = *(const v8h*)&vt[(nt * 16 + l16) * 32 + half * 8];
      v8h b1 = *(const v8h*)&vt[(nt * 16 + l16) * 32 + 16 + half * 8];
      #pragma unroll
      for (int e = 0; e < 8; ++e) { bv[e] = b0[e]; bv[8 + e] = b1[e]; }
      oc[nt] = wmma16(ap, bv, oc[nt]);
    }
  }

  #pragma unroll
  for (int nt = 0; nt < 4; ++nt)
    #pragma unroll
    for (int vv = 0; vv < 8; ++vv) {
      int qi = qt * 16 + half * 8 + vv;
      o[(((size_t)bb * SEQ + qi) * NHEAD + hh) * HEADD + nt * 16 + l16] = (_Float16)oc[nt][vv];
    }
}

// ---------------------------------------------------------------- launcher
extern "C" void kernel_launch(void* const* d_in, const int* in_sizes, int n_in,
                              void* d_out, int out_size, void* d_ws, size_t ws_size,
                              hipStream_t stream) {
  const int*   tokens = (const int*)d_in[0];
  const float* wte    = (const float*)d_in[1];
  const float* ln1_s  = (const float*)d_in[2];
  const float* ln1_b  = (const float*)d_in[3];
  const float* Wq     = (const float*)d_in[4];
  const float* bq     = (const float*)d_in[5];
  const float* Wk     = (const float*)d_in[6];
  const float* bk     = (const float*)d_in[7];
  const float* Wv     = (const float*)d_in[8];
  const float* bv     = (const float*)d_in[9];
  const float* Wo     = (const float*)d_in[10];
  const float* bo     = (const float*)d_in[11];
  const float* ln2_s  = (const float*)d_in[12];
  const float* ln2_b  = (const float*)d_in[13];
  const float* Wfi    = (const float*)d_in[14];
  const float* bfi    = (const float*)d_in[15];
  const float* Wfo    = (const float*)d_in[16];
  const float* bfo    = (const float*)d_in[17];
  const float* lnf_s  = (const float*)d_in[18];
  const float* lnf_b  = (const float*)d_in[19];
  float* out = (float*)d_out;

  const int BT = BATCH * SEQ;
  const int D = DMODEL, F = FFDIM, V = VOCAB;
  const size_t DD2 = (size_t)D * D;       // elems per DxD weight
  const size_t DF2 = (size_t)D * F;

  char* p = (char*)d_ws;
  char* pbase = p;
  auto carve = [&](size_t bytes) { char* r = p; p += (bytes + 255) & ~(size_t)255; return r; };
  float*    x    = (float*)carve((size_t)BT * D * 4);
  _Float16* h16  = (_Float16*)carve((size_t)BT * D * 2);   // LN out / attn out
  _Float16* q16  = (_Float16*)carve((size_t)BT * D * 2);
  _Float16* k16  = (_Float16*)carve((size_t)BT * D * 2);
  _Float16* v16  = (_Float16*)carve((size_t)BT * D * 2);
  _Float16* fb16 = (_Float16*)carve((size_t)BT * F * 2);
  _Float16* wbuf = (_Float16*)carve((4 * DD2 + 2 * DF2) * 2);  // per-layer f16 weights
  _Float16* wq16 = wbuf;
  _Float16* wk16 = wbuf + DD2;
  _Float16* wv16 = wbuf + 2 * DD2;
  _Float16* wo16 = wbuf + 3 * DD2;
  _Float16* wfi16 = wbuf + 4 * DD2;
  _Float16* wfo16 = wbuf + 4 * DD2 + DF2;
  size_t wteBytes = ((size_t)V * D * 2 + 255) & ~(size_t)255;
  bool haveWte16 = ((size_t)(p - pbase) + wteBytes) <= ws_size;
  _Float16* wte16 = haveWte16 ? (_Float16*)carve((size_t)V * D * 2) : nullptr;

  (void)hipFuncSetAttribute((const void*)attn_kernel,
                            hipFuncAttributeMaxDynamicSharedMemorySize, ATTN_SMEM);

  embed_kernel<<<BT, 256, 0, stream>>>(tokens, wte, x, D);
  if (haveWte16)
    conv16<<<4096, 256, 0, stream>>>(wte, wte16, (size_t)V * D);

  dim3 gDD(BT / 128, D / 128);
  dim3 gDF(BT / 128, F / 128);
  dim3 gDV(BT / 128, (V + 127) / 128);
  dim3 gAT(SEQ / 16, NHEAD, BATCH);
  dim3 tDD(D / 32, D / 32);
  dim3 tDF(D / 32, F / 32);
  dim3 tFD(F / 32, D / 32);

  for (int i = 0; i < N_LAYERS; ++i) {
    // convert+transpose this layer's weights to f16 [N][K]
    wconv_t<<<tDD, 256, 0, stream>>>(Wq + (size_t)i * DD2, wq16, D, D);
    wconv_t<<<tDD, 256, 0, stream>>>(Wk + (size_t)i * DD2, wk16, D, D);
    wconv_t<<<tDD, 256, 0, stream>>>(Wv + (size_t)i * DD2, wv16, D, D);
    wconv_t<<<tDD, 256, 0, stream>>>(Wo + (size_t)i * DD2, wo16, D, D);
    wconv_t<<<tDF, 256, 0, stream>>>(Wfi + (size_t)i * DF2, wfi16, D, F);
    wconv_t<<<tFD, 256, 0, stream>>>(Wfo + (size_t)i * DF2, wfo16, F, D);

    ln_kernel<<<BT, 256, 0, stream>>>(x, ln1_s + (size_t)i * D, ln1_b + (size_t)i * D, h16, D);
    gemm_h16<0><<<gDD, 256, 0, stream>>>(h16, wq16, bq + (size_t)i * D, q16, nullptr, BT, D, D, D);
    gemm_h16<0><<<gDD, 256, 0, stream>>>(h16, wk16, bk + (size_t)i * D, k16, nullptr, BT, D, D, D);
    gemm_h16<0><<<gDD, 256, 0, stream>>>(h16, wv16, bv + (size_t)i * D, v16, nullptr, BT, D, D, D);
    attn_kernel<<<gAT, 32, ATTN_SMEM, stream>>>(q16, k16, v16, h16);
    gemm_h16<2><<<gDD, 256, 0, stream>>>(h16, wo16, bo + (size_t)i * D, x, x, BT, D, D, D);
    ln_kernel<<<BT, 256, 0, stream>>>(x, ln2_s + (size_t)i * D, ln2_b + (size_t)i * D, h16, D);
    gemm_h16<1><<<gDF, 256, 0, stream>>>(h16, wfi16, bfi + (size_t)i * F, fb16, nullptr, BT, F, D, F);
    gemm_h16<2><<<gDD, 256, 0, stream>>>(fb16, wfo16, bfo + (size_t)i * D, x, x, BT, D, F, D);
  }

  ln_kernel<<<BT, 256, 0, stream>>>(x, lnf_s, lnf_b, h16, D);
  if (haveWte16)
    gemm_h16<3><<<gDV, 256, 0, stream>>>(h16, wte16, nullptr, out, nullptr, BT, V, D, V);
  else
    gemm_logits_f32w<<<gDV, 256, 0, stream>>>(h16, wte, out, BT, V, D, V);
}
